// FPredict2_80556406604409
// MI455X (gfx1250) — compile-verified, matrix operand order checked by reading, not run on hardware
//
#include <hip/hip_runtime.h>
#include <stdint.h>

// Problem constants (from reference)
#define BB     16
#define NC     80
#define HH     128
#define WW     128
#define HW_    (HH * WW)        // 16384
#define TOPK_  100
#define NMS_TH 0.45f
#define PAD_N  112              // 100 padded to 7*16 for WMMA tiling

typedef __attribute__((ext_vector_type(16))) _Float16 v16h;
typedef __attribute__((ext_vector_type(8)))  float    v8f;

__device__ __forceinline__ float sigmoidf_(float x) {
    return 1.0f / (1.0f + expf(-x));
}

// ---------------------------------------------------------------------------
// Kernel A: one block per (b,c) plane. Stage the 64KB plane into LDS with
// coalesced B128 loads, 5x5 peak mask on raw logits (sigmoid is monotonic, so
// peaks of sigmoid == peaks of logits -> transcendental only at peaks), then
// fused on-chip top-100 selection (no 84MB masked-score round trip to HBM).
// grid = B*NC blocks of 256 threads; dynamic LDS = 2*HW floats + scratch.
// ---------------------------------------------------------------------------
__global__ void peak_topk_kernel(const float* __restrict__ cls,
                                 float* __restrict__ tk_scores,
                                 int*   __restrict__ tk_inds) {
    extern __shared__ float smem[];
    float* raw   = smem;                 // HW_ floats (raw logits)
    float* score = smem + HW_;           // HW_ floats (masked sigmoid scores)
    float* redv  = smem + 2 * HW_;       // 256 floats
    int*   redi  = (int*)(smem + 2 * HW_ + 256); // 256 ints

    const int   plane = blockIdx.x;      // b*NC + c
    const float* src  = cls + (size_t)plane * HW_;
    const int   t     = threadIdx.x;

    // ---- stage plane Global -> LDS with vectorized B128 loads ----
    {
        const float4* s4 = (const float4*)src;
        float4*       d4 = (float4*)raw;
        for (int q = t; q < HW_ / 4; q += 256) d4[q] = s4[q];
    }
    __syncthreads();

    // ---- 5x5 peak mask on raw logits; sigmoid only at peaks ----
    for (int p = t; p < HW_; p += 256) {
        int x = p & (WW - 1), y = p >> 7;
        float v = raw[p];
        int y0 = y - 2 < 0 ? 0 : y - 2;
        int y1 = y + 2 > HH - 1 ? HH - 1 : y + 2;
        int x0 = x - 2 < 0 ? 0 : x - 2;
        int x1 = x + 2 > WW - 1 ? WW - 1 : x + 2;
        float m = -3.0e38f;
        for (int yy = y0; yy <= y1; ++yy)
            for (int xx = x0; xx <= x1; ++xx)
                m = fmaxf(m, raw[yy * WW + xx]);
        score[p] = (v == m) ? sigmoidf_(v) : 0.0f;
    }
    __syncthreads();

    // ---- iterative top-100 extraction (scores >= 0; -1 marks removed) ----
    float* outs = tk_scores + (size_t)plane * TOPK_;
    int*   outi = tk_inds   + (size_t)plane * TOPK_;
    for (int k = 0; k < TOPK_; ++k) {
        float bv = -2.0f; int bi = 0x7FFFFFFF;
        for (int p = t; p < HW_; p += 256) {
            float v = score[p];
            if (v > bv || (v == bv && p < bi)) { bv = v; bi = p; }
        }
        redv[t] = bv; redi[t] = bi;
        __syncthreads();
        for (int s = 128; s > 0; s >>= 1) {
            if (t < s) {
                float ov = redv[t + s]; int oi = redi[t + s];
                if (ov > redv[t] || (ov == redv[t] && oi < redi[t])) {
                    redv[t] = ov; redi[t] = oi;
                }
            }
            __syncthreads();
        }
        if (t == 0) {
            outs[k] = redv[0];
            outi[k] = redi[0];
            score[redi[0]] = -1.0f;
        }
        __syncthreads();
    }
}

// ---------------------------------------------------------------------------
// Kernel B: per batch — merge 80*100 candidates to final top-100.
// Writes s2 and clses straight into d_out; (cls, hw-index) ints to workspace.
// ---------------------------------------------------------------------------
__global__ void merge_topk_kernel(const float* __restrict__ tk_scores,
                                  const int*   __restrict__ tk_inds,
                                  float* __restrict__ out,
                                  int*   __restrict__ m_inds,
                                  int*   __restrict__ m_cls) {
    __shared__ float vals[NC * TOPK_];
    __shared__ float redv[256];
    __shared__ int   redi[256];
    const int b = blockIdx.x, t = threadIdx.x;

    for (int s = t; s < NC * TOPK_; s += 256) {
        __builtin_prefetch(&tk_scores[(size_t)b * NC * TOPK_ + s + 256], 0, 1);
        vals[s] = tk_scores[(size_t)b * NC * TOPK_ + s];
    }
    __syncthreads();

    float* s2 = out + BB * TOPK_ * 4 + b * TOPK_;
    float* cf = out + BB * TOPK_ * 4 + BB * TOPK_ + b * TOPK_;

    for (int k = 0; k < TOPK_; ++k) {
        float bv = -2.0f; int bi = 0x7FFFFFFF;
        for (int s = t; s < NC * TOPK_; s += 256) {
            float v = vals[s];
            if (v > bv || (v == bv && s < bi)) { bv = v; bi = s; }
        }
        redv[t] = bv; redi[t] = bi;
        __syncthreads();
        for (int s = 128; s > 0; s >>= 1) {
            if (t < s) {
                float ov = redv[t + s]; int oi = redi[t + s];
                if (ov > redv[t] || (ov == redv[t] && oi < redi[t])) {
                    redv[t] = ov; redi[t] = oi;
                }
            }
            __syncthreads();
        }
        if (t == 0) {
            int slot = redi[0];
            int c = slot / TOPK_;
            s2[k] = redv[0];
            cf[k] = (float)c;
            m_cls[b * TOPK_ + k]  = c;
            m_inds[b * TOPK_ + k] = tk_inds[(size_t)b * NC * TOPK_ + slot];
            vals[slot] = -1.0f;
        }
        __syncthreads();
    }
}

// ---------------------------------------------------------------------------
// Kernel C: per batch — decode 100 boxes, NMS.
// Pairwise union term area_i + area_j computed as a rank-1 outer sum with
// v_wmma_f32_16x16x32_f16 using a hi/lo f16 split of each area (error ~2^-22,
// accumulated in f32): A row m = [hi_m, lo_m, 1, 1, 0...], B col n =
// [1, 1, hi_n, lo_n, 0...]. Intersection + IoU compare done in VALU on the
// WMMA C/D lane layout. grid = B x 128 threads (4 waves over 49 tiles).
// ---------------------------------------------------------------------------
__global__ void decode_nms_kernel(const float* __restrict__ txty,
                                  const float* __restrict__ twth,
                                  const int*   __restrict__ m_inds,
                                  const int*   __restrict__ m_cls,
                                  float* __restrict__ out) {
    __shared__ float ob[PAD_N * 4];          // class-offset boxes
    __shared__ float area[PAD_N];
    __shared__ unsigned char sup[PAD_N * PAD_N];
    const int b = blockIdx.x, t = threadIdx.x;

    if (t < PAD_N) {
        float x1 = 0.f, y1 = 0.f, x2 = 0.f, y2 = 0.f;
        if (t < TOPK_) {
            int idx = m_inds[b * TOPK_ + t];
            int cls = m_cls[b * TOPK_ + t];
            int gx = idx & (WW - 1), gy = idx >> 7;
            const float* tp = txty + (size_t)b * 2 * HW_;
            const float* wp = twth + (size_t)b * 2 * HW_;
            float cx = (sigmoidf_(tp[idx]) + (float)gx) * 4.0f;
            float cy = (sigmoidf_(tp[HW_ + idx]) + (float)gy) * 4.0f;
            float hw = expf(wp[idx])       * 4.0f * 0.5f;
            float hh = expf(wp[HW_ + idx]) * 4.0f * 0.5f;
            x1 = fminf(fmaxf((cx - hw) * (1.0f / 512.0f), 0.f), 1.f);
            y1 = fminf(fmaxf((cy - hh) * (1.0f / 512.0f), 0.f), 1.f);
            x2 = fminf(fmaxf((cx + hw) * (1.0f / 512.0f), 0.f), 1.f);
            y2 = fminf(fmaxf((cy + hh) * (1.0f / 512.0f), 0.f), 1.f);
            float* bo = out + (b * TOPK_ + t) * 4;
            bo[0] = x1; bo[1] = y1; bo[2] = x2; bo[3] = y2;
            float off = 2.0f * (float)cls;
            x1 += off; y1 += off; x2 += off; y2 += off;
        }
        ob[t * 4 + 0] = x1; ob[t * 4 + 1] = y1;
        ob[t * 4 + 2] = x2; ob[t * 4 + 3] = y2;
        area[t] = (x2 - x1) * (y2 - y1);
    }
    __syncthreads();

    // 7x7 tiles of 16x16; 4 waves round-robin the 49 tiles
    const int wave = t >> 5, lane = t & 31;
    const int lr = lane & 15;
    const bool lolane = (lane < 16);
    const _Float16 one_h = (_Float16)1.0f;
    for (int tile = wave; tile < 49; tile += 4) {
        int m0 = (tile / 7) * 16, n0 = (tile % 7) * 16;
        v16h a = {};   // lanes 0-15 hold K=0..3 in first 4 halves; hi lanes K>=8 -> 0
        v16h bm = {};
        if (lolane) {
            float am = area[m0 + lr];
            _Float16 ah = (_Float16)am;
            _Float16 al = (_Float16)(am - (float)ah);
            a[0] = ah; a[1] = al; a[2] = one_h; a[3] = one_h;
            float an = area[n0 + lr];
            _Float16 bh = (_Float16)an;
            _Float16 bl = (_Float16)(an - (float)bh);
            bm[0] = one_h; bm[1] = one_h; bm[2] = bh; bm[3] = bl;
        }
        v8f c = {};
        v8f d = __builtin_amdgcn_wmma_f32_16x16x32_f16(
            false, a, false, bm, (short)0, c, false, false);
        // D[m][n] ~= area[m] + area[n]; lane n = lr, rows mb..mb+7
        int n  = n0 + lr;
        int mb = m0 + ((lane >> 4) << 3);
        float nx1 = ob[n * 4 + 0], ny1 = ob[n * 4 + 1];
        float nx2 = ob[n * 4 + 2], ny2 = ob[n * 4 + 3];
#pragma unroll
        for (int v = 0; v < 8; ++v) {
            int m = mb + v;
            float mx1 = ob[m * 4 + 0], my1 = ob[m * 4 + 1];
            float mx2 = ob[m * 4 + 2], my2 = ob[m * 4 + 3];
            float w  = fmaxf(1e-28f, fminf(mx2, nx2) - fmaxf(mx1, nx1));
            float h  = fmaxf(1e-28f, fminf(my2, ny2) - fmaxf(my1, ny1));
            float it = w * h;
            float iou = it / (d[v] - it);
            if (m < TOPK_ && n < TOPK_)
                sup[m * PAD_N + n] = (iou > NMS_TH) ? (unsigned char)1 : (unsigned char)0;
        }
    }
    __syncthreads();

    if (t == 0) {
        bool keep[TOPK_];
        for (int i = 0; i < TOPK_; ++i) keep[i] = true;
        for (int i = 0; i < TOPK_; ++i)
            if (keep[i])
                for (int j = i + 1; j < TOPK_; ++j)
                    if (sup[i * PAD_N + j]) keep[j] = false;
        float* ko = out + BB * TOPK_ * 4 + 2 * BB * TOPK_ + b * TOPK_;
        for (int i = 0; i < TOPK_; ++i) ko[i] = keep[i] ? 1.0f : 0.0f;
    }
}

// ---------------------------------------------------------------------------
extern "C" void kernel_launch(void* const* d_in, const int* in_sizes, int n_in,
                              void* d_out, int out_size, void* d_ws, size_t ws_size,
                              hipStream_t stream) {
    const float* cls  = (const float*)d_in[0];   // B,80,128,128
    const float* txty = (const float*)d_in[1];   // B,2,128,128
    const float* twth = (const float*)d_in[2];   // B,2,128,128
    float* out = (float*)d_out;

    char* ws = (char*)d_ws;
    float* tk_scores = (float*)ws;                                   // B*NC*TOPK f32
    int*   tk_inds   = (int*)(ws + (size_t)BB * NC * TOPK_ * 4);     // B*NC*TOPK i32
    int*   m_inds    = (int*)(ws + (size_t)2 * BB * NC * TOPK_ * 4); // B*TOPK i32
    int*   m_cls     = m_inds + BB * TOPK_;                          // B*TOPK i32

    const size_t smemA = (size_t)(2 * HW_ + 256) * sizeof(float) + 256 * sizeof(int);

    peak_topk_kernel<<<BB * NC, 256, smemA, stream>>>(cls, tk_scores, tk_inds);
    merge_topk_kernel<<<BB, 256, 0, stream>>>(tk_scores, tk_inds, out, m_inds, m_cls);
    decode_nms_kernel<<<BB, 128, 0, stream>>>(txty, twth, m_inds, m_cls, out);
}
